// AttentionV2_82119774699928
// MI455X (gfx1250) — matrix operand order repaired and checked by reading, not compiled, not run on hardware
//
#include <hip/hip_runtime.h>
#include <stdint.h>

// ---------------------------------------------------------------------------
// MI455X (gfx1250) attention block, wave32 + v_wmma_f32_16x16x32_f16 + TDM.
// Sizes: B=8, CIN=256, CH=128, HW=4096.
// Workspace layout (needs ~24.3 MB):
//   Qh  [B][HW][CH] f16   (c fastest)      8 MB
//   Kh  [B][HW][CH] f16   (c fastest)      8 MB
//   Vh  [B][CH][HW] f16   (p fastest)      8 MB
//   m   [B][HW]     f32   row max        128 KB
//   rl  [B][HW]     f32   1/row-sum      128 KB
// ---------------------------------------------------------------------------

typedef _Float16 half_t;
typedef __attribute__((ext_vector_type(16))) _Float16 v16h;
typedef __attribute__((ext_vector_type(8)))  float    v8f;

typedef unsigned int u32x4 __attribute__((ext_vector_type(4)));
typedef int          i32x4 __attribute__((ext_vector_type(4)));
typedef int          i32x8 __attribute__((ext_vector_type(8)));

#define NB   8
#define NC   256
#define NCH  128
#define NHW  4096

#if defined(__has_builtin)
#if __has_builtin(__builtin_amdgcn_tensor_load_to_lds) && \
    __has_builtin(__builtin_amdgcn_s_wait_tensorcnt)
#define HAVE_TDM 1
#endif
#endif
#ifndef HAVE_TDM
#define HAVE_TDM 0
#endif

union U16h { v16h v; uint32_t u[8]; };

__device__ __forceinline__ uint32_t pack2(float a, float b) {
  union { uint32_t u; _Float16 h[2]; } t;
  t.h[0] = (_Float16)a; t.h[1] = (_Float16)b;
  return t.u;
}

// A-operand dword->K order this code uses (contiguous, same as B): dword t, half-wave hi:
//   K = 2t + 16*hi
__device__ __forceinline__ int a_k(int t, int hi) {
  return (t << 1) + (hi << 4);
}
// B-operand dword->K order this code uses: K = 2t + 16*hi (N = lane&15)

__device__ __forceinline__ v8f wmma_f16(v16h a, v16h b, v8f c) {
  return __builtin_amdgcn_wmma_f32_16x16x32_f16(
      false, a, false, b, (short)0, c, false, false);
}

__device__ __forceinline__ v8f vzero8() {
  v8f z;
#pragma unroll
  for (int r = 0; r < 8; ++r) z[r] = 0.0f;
  return z;
}

// ---------------------------------------------------------------------------
// Kernel 1: projection  dst = W(128x256) @ src_tile(256 x 128p) + bias  -> f16
// grid (HW/128, B), block 256. layout_cp==0 -> [b][p][c], ==1 -> [b][c][p]
// TDM stages the f32 tile into LDS; one cooperative pass converts/transposes
// it to packed f16 so WMMA fragments come from ds_load_b128.
// ---------------------------------------------------------------------------
__global__ __launch_bounds__(256) void proj_kernel(
    const float* __restrict__ src,   // [B][256][4096]
    const float* __restrict__ W,     // [128][256]
    const float* __restrict__ bias,  // [128]
    half_t* __restrict__ dst, int layout_cp) {
  const int p0 = blockIdx.x * 128;
  const int b  = blockIdx.y;
  const int tid = threadIdx.x, wave = tid >> 5, lane = tid & 31;
  const int hi = (lane >> 4) & 1, nn = lane & 15;
  const int o0 = wave * 16;

  const float* xb = src + (size_t)b * NC * NHW;
#if HAVE_TDM
  __shared__ __align__(16) float ldsXf[32 * 128];   // [c][p] f32, 16 KB (DMA dest)
#endif
  __shared__ __align__(16) half_t ldsXh[128 * 32];  // [p][c] f16, 8 KB
  const uint32_t* ldsXu = (const uint32_t*)ldsXh;

  v8f acc[8];
#pragma unroll
  for (int jt = 0; jt < 8; ++jt) acc[jt] = vzero8();

#if HAVE_TDM
  // Per ISA: low 32 bits of a generic pointer to LDS are the wave-relative
  // LDS byte address (aperture maps addr[31:0] -> LDS).
  const uint32_t lds_off = (uint32_t)(size_t)(void*)ldsXf;
#endif

  for (int c0 = 0; c0 < NC; c0 += 32) {
    __syncthreads();  // prior-iter LDS consumers done
#if HAVE_TDM
    if (wave == 0) {
      // Tensor DMA descriptor: 2D tile, 32 rows(c) x 128 cols(p), f32,
      // row stride 4096 elements; dest LDS contiguous ([c][p]).
      const unsigned long long ga =
          (unsigned long long)(size_t)(xb + (size_t)c0 * NHW + p0);
      u32x4 g0;
      g0[0] = 1u;                                   // count=1, user mode
      g0[1] = lds_off;                              // lds_addr (bytes)
      g0[2] = (unsigned)(ga & 0xFFFFFFFFu);         // global_addr[31:0]
      g0[3] = (unsigned)((ga >> 32) & 0x01FFFFFFu)  // global_addr[56:32]
              | (2u << 30);                         // type=2 ("image")
      i32x8 g1;
      g1[0] = (int)(2u << 16);                      // mask=0, data_size=4B
      g1[1] = (int)(4096u << 16);                   // tensor_dim0 lo16 (4096)
      g1[2] = (int)((4096u >> 16) | (256u << 16));  // dim0 hi | tensor_dim1 lo
      g1[3] = (int)((256u >> 16) | (128u << 16));   // dim1 hi | tile_dim0=128
      g1[4] = (int)32;                              // tile_dim1=32, tile_dim2=0
      g1[5] = (int)4096;                            // tensor_dim0_stride lo32
      g1[6] = 0;                                    // stride hi | dim1_stride
      g1[7] = 0;
      i32x4 g2; g2[0] = 0; g2[1] = 0; g2[2] = 0; g2[3] = 0;  // dims>2 unused
      i32x4 g3; g3[0] = 0; g3[1] = 0; g3[2] = 0; g3[3] = 0;
#if __clang_major__ >= 23
      i32x8 g4;
#pragma unroll
      for (int q = 0; q < 8; ++q) g4[q] = 0;
      __builtin_amdgcn_tensor_load_to_lds(g0, g1, g2, g3, g4, 0);
#else
      __builtin_amdgcn_tensor_load_to_lds(g0, g1, g2, g3, 0);
#endif
      __builtin_amdgcn_s_wait_tensorcnt(0);
    }
    __syncthreads();  // DMA'd f32 tile visible
    // Convert + transpose once: [c][p] f32 -> [p][c] f16 (16 elems/thread)
#pragma unroll
    for (int q = 0; q < 16; ++q) {
      int idx = q * 256 + tid;
      int pl = idx & 127, cl = idx >> 7;
      ldsXh[pl * 32 + cl] = (half_t)ldsXf[cl * 128 + pl];
    }
#else
    // Fallback: stage + convert straight from global.
#pragma unroll
    for (int q = 0; q < 16; ++q) {
      int idx = q * 256 + tid;
      int pl = idx & 127, cl = idx >> 7;
      ldsXh[pl * 32 + cl] = (half_t)xb[(size_t)(c0 + cl) * NHW + p0 + pl];
    }
#endif
    __syncthreads();  // packed f16 tile ready

    // A = W[o0..o0+15][c0..c0+31] (f32 -> f16 on the fly)
    U16h A;
#pragma unroll
    for (int t = 0; t < 8; ++t) {
      int k = a_k(t, hi);
      float f0 = W[(o0 + nn) * NC + c0 + k];
      float f1 = W[(o0 + nn) * NC + c0 + k + 1];
      A.u[t] = pack2(f0, f1);
    }
#pragma unroll
    for (int jt = 0; jt < 8; ++jt) {
      U16h Bm;
#pragma unroll
      for (int t = 0; t < 8; ++t)
        Bm.u[t] = ldsXu[(jt * 16 + nn) * 16 + t + 8 * hi];
      acc[jt] = wmma_f16(A.v, Bm.v, acc[jt]);
    }
  }

  float bv[8];
#pragma unroll
  for (int r = 0; r < 8; ++r) bv[r] = bias[o0 + hi * 8 + r];

#pragma unroll
  for (int jt = 0; jt < 8; ++jt) {
    int p = p0 + jt * 16 + nn;
    if (layout_cp == 0) {  // [b][p][c]: 8 contiguous halves per lane
      union { uint32_t u4[4]; half_t h[8]; } pk;
#pragma unroll
      for (int r = 0; r < 8; ++r) pk.h[r] = (half_t)(acc[jt][r] + bv[r]);
      uint4 st; st.x = pk.u4[0]; st.y = pk.u4[1]; st.z = pk.u4[2]; st.w = pk.u4[3];
      *reinterpret_cast<uint4*>(dst + ((size_t)b * NHW + p) * NCH + o0 + hi * 8) = st;
    } else {               // [b][c][p]
#pragma unroll
      for (int r = 0; r < 8; ++r) {
        int o = o0 + hi * 8 + r;
        dst[((size_t)b * NCH + o) * NHW + p] = (half_t)(acc[jt][r] + bv[r]);
      }
    }
  }
}

// ---------------------------------------------------------------------------
// Kernel 2: softmax row stats over j of S = K^T Q.  grid (HW/16, B), block 256.
// Each wave covers 512 j columns; online (max,sum), shfl + LDS merge.
// ---------------------------------------------------------------------------
__global__ __launch_bounds__(256) void stats_kernel(
    const half_t* __restrict__ Qh, const half_t* __restrict__ Kh,
    float* __restrict__ mOut, float* __restrict__ rlOut) {
  const int b = blockIdx.y;
  const int i0 = blockIdx.x * 16;
  const int tid = threadIdx.x, wave = tid >> 5, lane = tid & 31;
  const int hi = (lane >> 4) & 1, nn = lane & 15;

  const uint32_t* Ku = (const uint32_t*)(Kh + (size_t)b * NHW * NCH);
  const uint32_t* Qu = (const uint32_t*)(Qh + (size_t)b * NHW * NCH);

  // Preload A = K^T[i0..i0+15][c], 4 chunks of 32 channels
  U16h A[4];
#pragma unroll
  for (int cc = 0; cc < 4; ++cc)
#pragma unroll
    for (int t = 0; t < 8; ++t) {
      int k = a_k(t, hi);
      A[cc].u[t] = Ku[(size_t)(i0 + nn) * 64 + ((cc * 32 + k) >> 1)];
    }

  float m[8], l[8];
#pragma unroll
  for (int r = 0; r < 8; ++r) { m[r] = -__builtin_inff(); l[r] = 0.0f; }

  const int j0 = wave * 512;
  for (int jt = 0; jt < 32; ++jt) {
    int j = j0 + jt * 16 + nn;
    v8f S = vzero8();
#pragma unroll
    for (int cc = 0; cc < 4; ++cc) {
      U16h Bm;
#pragma unroll
      for (int t = 0; t < 8; ++t)
        Bm.u[t] = Qu[(size_t)j * 64 + cc * 16 + t + 8 * hi];
      S = wmma_f16(A[cc].v, Bm.v, S);
    }
#pragma unroll
    for (int r = 0; r < 8; ++r) {
      float s = S[r];
      float mn = fmaxf(m[r], s);
      l[r] = l[r] * __expf(m[r] - mn) + __expf(s - mn);
      m[r] = mn;
    }
  }

  // merge across the 16 lanes of each half-wave
#pragma unroll
  for (int off = 1; off < 16; off <<= 1) {
#pragma unroll
    for (int r = 0; r < 8; ++r) {
      float m2 = __shfl_xor(m[r], off, 16);
      float l2 = __shfl_xor(l[r], off, 16);
      float mn = fmaxf(m[r], m2);
      l[r] = l[r] * __expf(m[r] - mn) + l2 * __expf(m2 - mn);
      m[r] = mn;
    }
  }

  __shared__ float sm[8][16], sl[8][16];
  if (nn == 0) {
#pragma unroll
    for (int r = 0; r < 8; ++r) {
      int row = hi * 8 + r;
      sm[wave][row] = m[r]; sl[wave][row] = l[r];
    }
  }
  __syncthreads();
  if (tid < 16) {
    float M = -__builtin_inff(), L = 0.0f;
#pragma unroll
    for (int w = 0; w < 8; ++w) {
      float m2 = sm[w][tid], l2 = sl[w][tid];
      float mn = fmaxf(M, m2);
      L = L * __expf(M - mn) + l2 * __expf(m2 - mn);
      M = mn;
    }
    mOut[(size_t)b * NHW + i0 + tid] = M;
    rlOut[(size_t)b * NHW + i0 + tid] = 1.0f / L;
  }
}

// ---------------------------------------------------------------------------
// Kernel 3: Y = V @ softmax(K^T Q) (recompute S), then out = Wz@Y + bz + x.
// grid (HW/128, B), block 256. Each wave owns a 16-channel strip of Y.
// ---------------------------------------------------------------------------
__global__ __launch_bounds__(256) void attn_out_kernel(
    const half_t* __restrict__ Qh, const half_t* __restrict__ Kh,
    const half_t* __restrict__ Vh,
    const float* __restrict__ mArr, const float* __restrict__ rlArr,
    const float* __restrict__ Wz, const float* __restrict__ bz,
    const float* __restrict__ x, float* __restrict__ out) {
  const int b = blockIdx.y;
  const int j0 = blockIdx.x * 128;
  const int tid = threadIdx.x, wave = tid >> 5, lane = tid & 31;
  const int hi = (lane >> 4) & 1, nn = lane & 15;

  const uint32_t* Ku = (const uint32_t*)(Kh + (size_t)b * NHW * NCH);
  const uint32_t* Qu = (const uint32_t*)(Qh + (size_t)b * NHW * NCH);
  const uint32_t* Vu = (const uint32_t*)(Vh + (size_t)b * NCH * NHW);
  const float* mB  = mArr  + (size_t)b * NHW;
  const float* rlB = rlArr + (size_t)b * NHW;

  __shared__ __align__(16) half_t ldsP[128 * 32];   // P tile [j][i], 8 KB
  __shared__ __align__(16) half_t ldsY[128 * 128];  // Y [j][ch], 32 KB
  uint32_t* Pu = (uint32_t*)ldsP;
  const uint32_t* Puc = (const uint32_t*)ldsP;
  uint32_t* Yu = (uint32_t*)ldsY;
  const uint32_t* Yuc = (const uint32_t*)ldsY;

  const int jloc = wave * 16 + nn;

  // Preload B = Q[c][j0+jloc] chunks (fixed for whole i-loop)
  U16h QB[4];
#pragma unroll
  for (int cc = 0; cc < 4; ++cc)
#pragma unroll
    for (int t = 0; t < 8; ++t)
      QB[cc].u[t] = Qu[(size_t)(j0 + jloc) * 64 + cc * 16 + t + 8 * hi];

  v8f Y[8];
#pragma unroll
  for (int jt = 0; jt < 8; ++jt) Y[jt] = vzero8();

  for (int i0 = 0; i0 < NHW; i0 += 32) {
    __builtin_prefetch(Ku + (size_t)(i0 + 32 + nn) * 64, 0, 1);  // next K tile
    // --- recompute S tile (32 i x 16 j per wave) ---
    v8f St = vzero8(), Sb = vzero8();
#pragma unroll
    for (int cc = 0; cc < 4; ++cc) {
      U16h At, Ab;
#pragma unroll
      for (int t = 0; t < 8; ++t) {
        int kh = (a_k(t, hi) >> 1);
        At.u[t] = Ku[(size_t)(i0 + nn) * 64 + cc * 16 + kh];
        Ab.u[t] = Ku[(size_t)(i0 + 16 + nn) * 64 + cc * 16 + kh];
      }
      St = wmma_f16(At.v, QB[cc].v, St);
      Sb = wmma_f16(Ab.v, QB[cc].v, Sb);
    }
    // --- P = exp(S - m) * (1/l), packed f16 into LDS [j][i] ---
#pragma unroll
    for (int r = 0; r < 8; r += 2) {
      int ir = hi * 8 + r;
      float p0 = __expf(St[r]     - mB[i0 + ir])     * rlB[i0 + ir];
      float p1 = __expf(St[r + 1] - mB[i0 + ir + 1]) * rlB[i0 + ir + 1];
      Pu[jloc * 16 + (ir >> 1)] = pack2(p0, p1);
      float q0 = __expf(Sb[r]     - mB[i0 + 16 + ir])     * rlB[i0 + 16 + ir];
      float q1 = __expf(Sb[r + 1] - mB[i0 + 16 + ir + 1]) * rlB[i0 + 16 + ir + 1];
      Pu[jloc * 16 + 8 + (ir >> 1)] = pack2(q0, q1);
    }
    __syncthreads();
    // --- Y += V[strip][i0..i0+31] @ P ---
    U16h VA;
#pragma unroll
    for (int t = 0; t < 8; ++t) {
      int k = a_k(t, hi);
      VA.u[t] = Vu[(size_t)(wave * 16 + nn) * 2048 + ((i0 + k) >> 1)];
    }
#pragma unroll
    for (int jt = 0; jt < 8; ++jt) {
      U16h Pb;
#pragma unroll
      for (int t = 0; t < 8; ++t)
        Pb.u[t] = Puc[(jt * 16 + nn) * 16 + t + 8 * hi];
      Y[jt] = wmma_f16(VA.v, Pb.v, Y[jt]);
    }
    __syncthreads();
  }

  // --- Y -> LDS [j][ch] (f16, ch-pairs contiguous) ---
#pragma unroll
  for (int jt = 0; jt < 8; ++jt) {
    int j = jt * 16 + nn;
#pragma unroll
    for (int r = 0; r < 8; r += 2) {
      int ch = wave * 16 + hi * 8 + r;
      Yu[j * 64 + (ch >> 1)] = pack2(Y[jt][r], Y[jt][r + 1]);
    }
  }
  __syncthreads();

  // --- Z = Wz(256x128) @ Y(128x128): wave owns o rows [32w, 32w+32) ---
  v8f Z0[8], Z1[8];
#pragma unroll
  for (int jt = 0; jt < 8; ++jt) { Z0[jt] = vzero8(); Z1[jt] = vzero8(); }
#pragma unroll
  for (int cc = 0; cc < 4; ++cc) {
    U16h W0, W1;
#pragma unroll
    for (int t = 0; t < 8; ++t) {
      int k = a_k(t, hi);
      int o = wave * 32 + nn;
      W0.u[t] = pack2(Wz[o * NCH + cc * 32 + k], Wz[o * NCH + cc * 32 + k + 1]);
      o += 16;
      W1.u[t] = pack2(Wz[o * NCH + cc * 32 + k], Wz[o * NCH + cc * 32 + k + 1]);
    }
#pragma unroll
    for (int jt = 0; jt < 8; ++jt) {
      U16h Yb;
#pragma unroll
      for (int t = 0; t < 8; ++t)
        Yb.u[t] = Yuc[(jt * 16 + nn) * 64 + cc * 16 + t + 8 * hi];
      Z0[jt] = wmma_f16(W0.v, Yb.v, Z0[jt]);
      Z1[jt] = wmma_f16(W1.v, Yb.v, Z1[jt]);
    }
  }

  // --- out = Z + bz + x (residual) ---
  float bz0[8], bz1[8];
#pragma unroll
  for (int r = 0; r < 8; ++r) {
    bz0[r] = bz[wave * 32 + hi * 8 + r];
    bz1[r] = bz[wave * 32 + 16 + hi * 8 + r];
  }
#pragma unroll
  for (int jt = 0; jt < 8; ++jt) {
    int j = j0 + jt * 16 + nn;
#pragma unroll
    for (int r = 0; r < 8; ++r) {
      int o = wave * 32 + hi * 8 + r;
      size_t idx = ((size_t)b * NC + o) * NHW + j;
      out[idx] = Z0[jt][r] + bz0[r] + x[idx];
      size_t idx2 = idx + (size_t)16 * NHW;
      out[idx2] = Z1[jt][r] + bz1[r] + x[idx2];
    }
  }
}

// ---------------------------------------------------------------------------
extern "C" void kernel_launch(void* const* d_in, const int* in_sizes, int n_in,
                              void* d_out, int out_size, void* d_ws, size_t ws_size,
                              hipStream_t stream) {
  (void)in_sizes; (void)n_in; (void)out_size; (void)ws_size;
  const float* x    = (const float*)d_in[0];
  const float* aux  = (const float*)d_in[1];
  const float* Wq_w = (const float*)d_in[2];
  const float* Wq_b = (const float*)d_in[3];
  const float* Wk_w = (const float*)d_in[4];
  const float* Wk_b = (const float*)d_in[5];
  const float* Wv_w = (const float*)d_in[6];
  const float* Wv_b = (const float*)d_in[7];
  const float* Wz_w = (const float*)d_in[8];
  const float* Wz_b = (const float*)d_in[9];
  float* out = (float*)d_out;

  const size_t QKV_BYTES  = (size_t)NB * NHW * NCH * sizeof(half_t);  // 8 MB
  const size_t STAT_BYTES = (size_t)NB * NHW * sizeof(float);         // 128 KB
  char* ws = (char*)d_ws;
  half_t* Qh = (half_t*)(ws);
  half_t* Kh = (half_t*)(ws + QKV_BYTES);
  half_t* Vh = (half_t*)(ws + 2 * QKV_BYTES);
  float*  mA = (float*)(ws + 3 * QKV_BYTES);
  float*  rl = (float*)(ws + 3 * QKV_BYTES + STAT_BYTES);

  dim3 blk(256);
  proj_kernel<<<dim3(NHW / 128, NB), blk, 0, stream>>>(x,   Wq_w, Wq_b, Qh, 0);
  proj_kernel<<<dim3(NHW / 128, NB), blk, 0, stream>>>(aux, Wk_w, Wk_b, Kh, 0);
  proj_kernel<<<dim3(NHW / 128, NB), blk, 0, stream>>>(aux, Wv_w, Wv_b, Vh, 1);
  stats_kernel<<<dim3(NHW / 16, NB), blk, 0, stream>>>(Qh, Kh, mA, rl);
  attn_out_kernel<<<dim3(NHW / 128, NB), blk, 0, stream>>>(
      Qh, Kh, Vh, mA, rl, Wz_w, Wz_b, x, out);
}
